// GQA_35296041239104
// MI455X (gfx1250) — matrix operand order repaired and checked by reading, not compiled, not run on hardware
//
#include <hip/hip_runtime.h>
#include <cmath>
#include <stdint.h>

#define EMB      2048
#define HEADS    16
#define GQA_G    2
#define HEAD_DIM 128
#define KVH      8
#define BATCH    2
#define SEQ      2048
#define MTOK     (BATCH * SEQ)

typedef __attribute__((ext_vector_type(16))) __bf16        v16bf;
typedef __attribute__((ext_vector_type(8)))  float         v8f;
typedef __attribute__((ext_vector_type(4)))  unsigned int  uint4v;

union Frag16 {
    v16bf  bf;
    uint4v q[2];
};

__device__ __forceinline__ unsigned short f2bf(float f) {
    unsigned u = __float_as_uint(f);
    u += 0x7FFFu + ((u >> 16) & 1u);   // round-to-nearest-even
    return (unsigned short)(u >> 16);
}

// ---------------------------------------------------------------------------
// fp32 -> bf16 elementwise convert
// ---------------------------------------------------------------------------
__global__ void f32_to_bf16_kernel(const float* __restrict__ src,
                                   unsigned short* __restrict__ dst, int n) {
    int i = blockIdx.x * blockDim.x + threadIdx.x;
    if (i < n) dst[i] = f2bf(src[i]);
}

// ---------------------------------------------------------------------------
// bf16 WMMA GEMM:  C[M,N] (f32) = A[M,K] (bf16, row-major) * W[N,K]^T (bf16)
// block = 128 threads (4 waves, 2x2); wave tile = 32x64 (2x4 frags, 8 acc)
// block tile = 64x128; grid = (N/128, M/64)
// Per 32-wide k-step: 12 x 16B loads per lane feed 8 WMMAs (21.3 FLOP/B).
// ---------------------------------------------------------------------------
__global__ void __launch_bounds__(128)
gemm_bf16_wmma(const unsigned short* __restrict__ A,
               const unsigned short* __restrict__ W,
               float* __restrict__ C, int N, int K) {
    const int lane = threadIdx.x & 31;
    const int wave = threadIdx.x >> 5;
    const int g    = lane >> 4;     // lane-half (0/1)
    const int ln   = lane & 15;
    const int wr   = wave >> 1;     // wave row (M)
    const int wc   = wave & 1;      // wave col (N)

    const int m0 = blockIdx.y * 64 + wr * 32;
    const int n0 = blockIdx.x * 128 + wc * 64;

    v8f acc[2][4];
#pragma unroll
    for (int mt = 0; mt < 2; ++mt)
#pragma unroll
        for (int nt = 0; nt < 4; ++nt)
            acc[mt][nt] = (v8f){0.f, 0.f, 0.f, 0.f, 0.f, 0.f, 0.f, 0.f};

    const unsigned short* arow[2];
#pragma unroll
    for (int mt = 0; mt < 2; ++mt)
        arow[mt] = A + (size_t)(m0 + mt * 16 + ln) * K;
    const unsigned short* wrow[4];
#pragma unroll
    for (int nt = 0; nt < 4; ++nt)
        wrow[nt] = W + (size_t)(n0 + nt * 16 + ln) * K;

    for (int k0 = 0; k0 < K; k0 += 32) {
        Frag16 a[2];
#pragma unroll
        for (int mt = 0; mt < 2; ++mt) {
            // A frag: elems 0..7 -> K = k0 + g*8 + i ; elems 8..15 -> +16
            a[mt].q[0] = *(const uint4v*)(arow[mt] + k0 + g * 8);
            a[mt].q[1] = *(const uint4v*)(arow[mt] + k0 + 16 + g * 8);
        }
#pragma unroll
        for (int nt = 0; nt < 4; ++nt) {
            // B frag: lane = col n, elems 0..7 -> K = k0 + g*8 + i ; elems 8..15 -> +16 (row of W = B^T)
            Frag16 b;
            b.q[0] = *(const uint4v*)(wrow[nt] + k0 + g * 8);
            b.q[1] = *(const uint4v*)(wrow[nt] + k0 + 16 + g * 8);
#pragma unroll
            for (int mt = 0; mt < 2; ++mt)
                acc[mt][nt] = __builtin_amdgcn_wmma_f32_16x16x32_bf16(
                    false, a[mt].bf, false, b.bf, (short)0, acc[mt][nt], false, false);
        }
    }

#pragma unroll
    for (int mt = 0; mt < 2; ++mt)
#pragma unroll
        for (int r = 0; r < 8; ++r) {      // C layout: VGPR r = row r + 8*g
            int row = m0 + mt * 16 + r + 8 * g;
#pragma unroll
            for (int nt = 0; nt < 4; ++nt)
                C[(size_t)row * N + n0 + nt * 16 + ln] = acc[mt][nt][r];
        }
}

// ---------------------------------------------------------------------------
// RoPE + scatter: src f32 [B,S, D*nh] (channel e = d*nh + hh) ->
//   bf16 dst [B,nh,S,D] (vtrans=0) or [B,nh,D,S] (vtrans=1)
// ---------------------------------------------------------------------------
__global__ void rope_scatter_kernel(const float* __restrict__ src,
                                    unsigned short* __restrict__ dst,
                                    int nh, int doRope, int vtrans, float scale) {
    int idx   = blockIdx.x * blockDim.x + threadIdx.x;
    int total = BATCH * SEQ * nh * HEAD_DIM;
    if (idx >= total) return;
    int d  = idx & (HEAD_DIM - 1);
    int hh = (idx >> 7) % nh;
    int s  = (idx / (HEAD_DIM * nh)) % SEQ;
    int b  = idx / (HEAD_DIM * nh * SEQ);

    const float* row = src + (size_t)(b * SEQ + s) * (nh * HEAD_DIM);
    float x = row[d * nh + hh];
    float val;
    if (doRope) {
        int   dm  = d & 63;
        float inv = __expf(-(float)dm * (9.210340371976184f / 64.0f)); // 10000^(-dm/64)
        float ang = (float)s * inv;
        float c = cosf(ang), sn = sinf(ang);
        int   dd    = (d < 64) ? (d + 64) : (d - 64);
        float other = row[dd * nh + hh];
        float rot   = (d < 64) ? -other : other;
        val = x * c + rot * sn;
    } else {
        val = x;
    }
    val *= scale;
    size_t o = vtrans
        ? (((size_t)(b * nh + hh)) * HEAD_DIM + d) * SEQ + s
        : (((size_t)(b * nh + hh)) * SEQ + s) * HEAD_DIM + d;
    dst[o] = f2bf(val);
}

// ---------------------------------------------------------------------------
// Flash attention: Q [B,H,S,D] bf16 (pre-scaled), K [B,KV,S,D] bf16,
// Vt [B,KV,D,S] bf16 -> ctx [B,S,H*D] bf16
// block = 128 threads (4 waves); wave = 16 query rows; grid = (S/64, B*H)
// ---------------------------------------------------------------------------
__global__ void __launch_bounds__(128)
attn_kernel(const unsigned short* __restrict__ Q,
            const unsigned short* __restrict__ Km,
            const unsigned short* __restrict__ Vt,
            unsigned short* __restrict__ ctx) {
    __shared__ unsigned short Plds[4][16][32];   // per-wave P staging (C->A relayout)

    const int lane = threadIdx.x & 31;
    const int wave = threadIdx.x >> 5;
    const int g    = lane >> 4;
    const int ln   = lane & 15;

    const int b  = blockIdx.y >> 4;
    const int h  = blockIdx.y & 15;
    const int kh = h / GQA_G;
    const int s0 = blockIdx.x * 64 + wave * 16;

    const unsigned short* Qb = Q  + ((size_t)(b * HEADS + h)) * SEQ * HEAD_DIM;
    const unsigned short* Kb = Km + ((size_t)(b * KVH + kh)) * SEQ * HEAD_DIM;
    const unsigned short* Vb = Vt + ((size_t)(b * KVH + kh)) * HEAD_DIM * SEQ;

    // Q fragments for 16 rows x 128 (4 k-steps of 32)
    Frag16 qf[4];
#pragma unroll
    for (int ks = 0; ks < 4; ++ks) {
        const unsigned short* qr = Qb + (size_t)(s0 + ln) * HEAD_DIM + ks * 32;
        qf[ks].q[0] = *(const uint4v*)(qr + g * 16);
        qf[ks].q[1] = *(const uint4v*)(qr + g * 16 + 8);
    }

    v8f zero = {0.f, 0.f, 0.f, 0.f, 0.f, 0.f, 0.f, 0.f};
    v8f acc[8];
#pragma unroll
    for (int dt = 0; dt < 8; ++dt) acc[dt] = zero;
    float mrow[8], lrow[8];
#pragma unroll
    for (int r = 0; r < 8; ++r) { mrow[r] = -1e30f; lrow[r] = 0.f; }

    for (int t0 = 0; t0 < SEQ; t0 += 32) {
        // ---- scores S[16 x 32] = Q . K^T (pre-scaled) ----
        v8f sc0 = zero, sc1 = zero;
#pragma unroll
        for (int ks = 0; ks < 4; ++ks) {
            Frag16 kf0, kf1;
            const unsigned short* kr0 = Kb + (size_t)(t0 + ln)      * HEAD_DIM + ks * 32 + g * 16;
            const unsigned short* kr1 = Kb + (size_t)(t0 + 16 + ln) * HEAD_DIM + ks * 32 + g * 16;
            kf0.q[0] = *(const uint4v*)(kr0);
            kf0.q[1] = *(const uint4v*)(kr0 + 8);
            kf1.q[0] = *(const uint4v*)(kr1);
            kf1.q[1] = *(const uint4v*)(kr1 + 8);
            sc0 = __builtin_amdgcn_wmma_f32_16x16x32_bf16(false, qf[ks].bf, false, kf0.bf,
                                                          (short)0, sc0, false, false);
            sc1 = __builtin_amdgcn_wmma_f32_16x16x32_bf16(false, qf[ks].bf, false, kf1.bf,
                                                          (short)0, sc1, false, false);
        }

        // ---- online softmax (rows live in lane-halves; reduce over 16 lanes) ----
#pragma unroll
        for (int r = 0; r < 8; ++r) {
            float mloc = fmaxf(sc0[r], sc1[r]);
#pragma unroll
            for (int m = 1; m < 16; m <<= 1)
                mloc = fmaxf(mloc, __shfl_xor(mloc, m, 32));
            float mnew = fmaxf(mrow[r], mloc);
            float p0 = __expf(sc0[r] - mnew);
            float p1 = __expf(sc1[r] - mnew);
            float rs = p0 + p1;
#pragma unroll
            for (int m = 1; m < 16; m <<= 1)
                rs += __shfl_xor(rs, m, 32);
            float corr = __expf(mrow[r] - mnew);
            lrow[r] = lrow[r] * corr + rs;
            mrow[r] = mnew;
#pragma unroll
            for (int dt = 0; dt < 8; ++dt) acc[dt][r] *= corr;
            int row = r + 8 * g;
            Plds[wave][row][ln]      = f2bf(p0);
            Plds[wave][row][16 + ln] = f2bf(p1);
        }
        __syncthreads();

        // P relayout: C-layout in LDS -> A-fragment (16x32)
        Frag16 pf;
        pf.q[0] = *(const uint4v*)(&Plds[wave][ln][g * 16]);
        pf.q[1] = *(const uint4v*)(&Plds[wave][ln][g * 16 + 8]);

        // ---- ctx += P . V  (V transposed: rows are d, contiguous in t) ----
#pragma unroll
        for (int dt = 0; dt < 8; ++dt) {
            Frag16 vf;
            const unsigned short* vr = Vb + (size_t)(dt * 16 + ln) * SEQ + t0 + g * 16;
            vf.q[0] = *(const uint4v*)(vr);
            vf.q[1] = *(const uint4v*)(vr + 8);
            acc[dt] = __builtin_amdgcn_wmma_f32_16x16x32_bf16(false, pf.bf, false, vf.bf,
                                                              (short)0, acc[dt], false, false);
        }
        __syncthreads();

        if (t0 + 32 < SEQ)
            __builtin_prefetch(Kb + (size_t)(t0 + 32 + ln) * HEAD_DIM, 0, 1);
    }

    // ---- normalize + store ctx (bf16), layout e = h*128 + d ----
#pragma unroll
    for (int r = 0; r < 8; ++r) {
        float invl = 1.0f / lrow[r];
        int   srow = s0 + r + 8 * g;
        unsigned short* crow = ctx + ((size_t)(b * SEQ + srow)) * EMB + h * HEAD_DIM;
#pragma unroll
        for (int dt = 0; dt < 8; ++dt)
            crow[dt * 16 + ln] = f2bf(acc[dt][r] * invl);
    }
}

// ---------------------------------------------------------------------------
extern "C" void kernel_launch(void* const* d_in, const int* in_sizes, int n_in,
                              void* d_out, int out_size, void* d_ws, size_t ws_size,
                              hipStream_t stream) {
    const float* x  = (const float*)d_in[0];
    const float* Wq = (const float*)d_in[1];
    const float* Wk = (const float*)d_in[2];
    const float* Wv = (const float*)d_in[3];
    const float* Wo = (const float*)d_in[4];
    float* out = (float*)d_out;

    // workspace layout (bf16 element counts; all multiples of 8 -> 16B aligned)
    constexpr size_t SZ_XB = (size_t)MTOK * EMB;
    constexpr size_t SZ_WQ = (size_t)EMB * EMB;
    constexpr size_t SZ_WK = (size_t)(EMB / GQA_G) * EMB;
    constexpr size_t SZ_WV = SZ_WK;
    constexpr size_t SZ_WO = SZ_WQ;
    constexpr size_t SZ_QF = (size_t)MTOK * EMB;              // f32
    constexpr size_t SZ_KF = (size_t)MTOK * (EMB / GQA_G);    // f32
    constexpr size_t SZ_VF = SZ_KF;                           // f32

    unsigned short* xb  = (unsigned short*)d_ws;
    unsigned short* wqb = xb  + SZ_XB;
    unsigned short* wkb = wqb + SZ_WQ;
    unsigned short* wvb = wkb + SZ_WK;
    unsigned short* wob = wvb + SZ_WV;
    float* Qf = (float*)(wob + SZ_WO);
    float* Kf = Qf + SZ_QF;
    float* Vf = Kf + SZ_KF;
    unsigned short* Qr  = (unsigned short*)(Vf + SZ_VF);
    unsigned short* Kr  = Qr + (size_t)MTOK * EMB;            // B*H*S*D
    unsigned short* Vt  = Kr + (size_t)MTOK * (EMB / GQA_G);  // B*KV*S*D
    unsigned short* ctx = (unsigned short*)Qf;                // alias (Qf dead after rope)

    const int T = 256;
    auto cdiv = [](int a, int b) { return (a + b - 1) / b; };

    f32_to_bf16_kernel<<<cdiv(MTOK * EMB, T), T, 0, stream>>>(x,  xb,  MTOK * EMB);
    f32_to_bf16_kernel<<<cdiv(EMB * EMB, T), T, 0, stream>>>(Wq, wqb, EMB * EMB);
    f32_to_bf16_kernel<<<cdiv(EMB * EMB / GQA_G, T), T, 0, stream>>>(Wk, wkb, EMB * EMB / GQA_G);
    f32_to_bf16_kernel<<<cdiv(EMB * EMB / GQA_G, T), T, 0, stream>>>(Wv, wvb, EMB * EMB / GQA_G);
    f32_to_bf16_kernel<<<cdiv(EMB * EMB, T), T, 0, stream>>>(Wo, wob, EMB * EMB);

    dim3 blk(128);
    // grid = (N/128, M/64)
    gemm_bf16_wmma<<<dim3(EMB / 128, MTOK / 64), blk, 0, stream>>>(xb, wqb, Qf, EMB, EMB);
    gemm_bf16_wmma<<<dim3((EMB / GQA_G) / 128, MTOK / 64), blk, 0, stream>>>(xb, wkb, Kf, EMB / GQA_G, EMB);
    gemm_bf16_wmma<<<dim3((EMB / GQA_G) / 128, MTOK / 64), blk, 0, stream>>>(xb, wvb, Vf, EMB / GQA_G, EMB);

    float qscale = 1.0f / sqrtf((float)EMB);
    rope_scatter_kernel<<<cdiv(BATCH * SEQ * HEADS * HEAD_DIM, T), T, 0, stream>>>(
        Qf, Qr, HEADS, 1, 0, qscale);
    rope_scatter_kernel<<<cdiv(BATCH * SEQ * KVH * HEAD_DIM, T), T, 0, stream>>>(
        Kf, Kr, KVH, 1, 0, 1.0f);
    rope_scatter_kernel<<<cdiv(BATCH * SEQ * KVH * HEAD_DIM, T), T, 0, stream>>>(
        Vf, Vt, KVH, 0, 1, 1.0f);

    attn_kernel<<<dim3(SEQ / 64, BATCH * HEADS), blk, 0, stream>>>(Qr, Kr, Vt, ctx);

    gemm_bf16_wmma<<<dim3(EMB / 128, MTOK / 64), blk, 0, stream>>>(ctx, wob, out, MTOK, EMB);
}
